// HRMEdgePredictor_59528246723280
// MI455X (gfx1250) — matrix operand order repaired and checked by reading, not compiled, not run on hardware
//
#include <hip/hip_runtime.h>

// ---------------------------------------------------------------------------
// HRM forward for MI455X (gfx1250, wave32, WMMA f32_16x16x32_f16,
// async global->LDS copies where data is already f16)
// ---------------------------------------------------------------------------

typedef _Float16 half_t;
typedef __attribute__((ext_vector_type(16))) _Float16 v16h;
typedef __attribute__((ext_vector_type(8)))  float    v8f;
typedef int v4i_ __attribute__((vector_size(16)));

#define B_     4
#define S_     1024
#define D_     1024
#define NH_    16
#define HD_    64
#define INTER_ 2816
#define M_     (B_*S_)

// ---------------- async global->LDS (gfx1250 ASYNCcnt path) -----------------

#if defined(__has_builtin)
#if __has_builtin(__builtin_amdgcn_global_load_async_to_lds_b128)
#define HAVE_ASYNC_LDS 1
#endif
#endif
#ifndef HAVE_ASYNC_LDS
#define HAVE_ASYNC_LDS 0
#endif

// Copy 16 bytes global -> LDS (ASYNCcnt-tracked, no VGPR round-trip).
// Falls back to a VGPR copy if the async builtin is unavailable.
__device__ __forceinline__ void async_cp16(void* ldst, const void* gsrc) {
#if HAVE_ASYNC_LDS
  __builtin_amdgcn_global_load_async_to_lds_b128(
      (__attribute__((address_space(1))) v4i_*)(unsigned long long)gsrc,
      (__attribute__((address_space(3))) v4i_*)(unsigned)(unsigned long long)ldst,
      0, 0);
#else
  *(float4*)ldst = *(const float4*)gsrc;
#endif
}

__device__ __forceinline__ void async_wait() {
#if HAVE_ASYNC_LDS
#if __has_builtin(__builtin_amdgcn_s_wait_asynccnt)
  __builtin_amdgcn_s_wait_asynccnt(0);
#else
  asm volatile("s_wait_asynccnt 0x0" ::: "memory");
#endif
#endif
}

// ------------------------------- WMMA core ---------------------------------

__device__ __forceinline__ v8f wmma16(v16h a, v16h b, v8f c) {
  // D = A(16x32 f16) * B(32x16 f16) + C(16x16 f32)
  return __builtin_amdgcn_wmma_f32_16x16x32_f16(false, a, false, b,
                                                (short)0, c, false, false);
}

// A operand: 16x32 tile, source row-major [M][K] in LDS (ld halfs, even).
// Lanes 0-15 = M with K=0..15 across v0..7; lanes 16-31 same M with K=16..31.
__device__ __forceinline__ v16h load_a_frag(const half_t* p, int ld) {
  const int lane = threadIdx.x & 31;
  const int l = lane & 15, hi = lane >> 4;
  union { v16h v; unsigned u[8]; } r;
  const half_t* row = p + l * ld;
#pragma unroll
  for (int j = 0; j < 8; ++j) {
    int kk = hi * 16 + 2 * j;
    r.u[j] = *(const unsigned*)(row + kk);
  }
  return r.v;
}

// B operand: 32x16 tile, source stored as [N][K] row-major (ld halfs, even).
// Lanes 0-15 = N with K=0..15 across v0..7; lanes 16-31 = N with K=16..31.
__device__ __forceinline__ v16h load_b_frag_nk(const half_t* p, int ld) {
  const int lane = threadIdx.x & 31;
  const int l = lane & 15, hi = lane >> 4;
  union { v16h v; unsigned u[8]; } r;
  const half_t* row = p + l * ld;
#pragma unroll
  for (int j = 0; j < 8; ++j) {
    int kk = hi * 16 + 2 * j;
    r.u[j] = *(const unsigned*)(row + kk);
  }
  return r.v;
}

// B operand from [K][N]-stored tile (e.g. V in attention): B[k][n] = p[k*ld+n].
__device__ __forceinline__ v16h load_b_frag_kn(const half_t* p, int ld) {
  const int lane = threadIdx.x & 31;
  const int l = lane & 15, hi = lane >> 4;
  union { v16h v; half_t h[16]; } r;
#pragma unroll
  for (int j = 0; j < 8; ++j) {
    int k0 = hi * 16 + 2 * j;
    r.h[2 * j]     = p[(size_t)k0 * ld + l];
    r.h[2 * j + 1] = p[(size_t)(k0 + 1) * ld + l];
  }
  return r.v;
}

// ------------------------------ GEMM: C = A * W^T ---------------------------
// A: [M,K] f32 row-major; W: [N,K] f32 row-major (torch [out,in]); C: [M,N].
// Block tile 128x128, BK=32; 8 waves (2Mx4N), each wave 64x32 (8 WMMA accum).

template <bool OUT_HALF>
__global__ __launch_bounds__(256) void gemm_xwt(
    const float* __restrict__ A, const float* __restrict__ W,
    void* __restrict__ Cv, int Mdim, int Ndim, int Kdim) {
  __shared__ half_t As[128 * 40];
  __shared__ half_t Bs[128 * 40];

  const int tid  = threadIdx.x;
  const int lane = tid & 31;
  const int wave = tid >> 5;
  const int l    = lane & 15;
  const int hi   = lane >> 4;
  const int wm   = (wave >> 2) * 64;   // 2 wave rows
  const int wn   = (wave & 3) * 32;    // 4 wave cols
  const int bm   = blockIdx.y * 128;
  const int bn   = blockIdx.x * 128;

  v8f acc[4][2] = {};

  for (int k0 = 0; k0 < Kdim; k0 += 32) {
#pragma unroll
    for (int i = 0; i < 4; ++i) {          // A tile: 128 x 32 f32 -> f16
      int idx = tid + 256 * i;
      int r = idx >> 3, c4 = idx & 7;
      float4 f = *(const float4*)&A[(size_t)(bm + r) * Kdim + k0 + c4 * 4];
      half_t* d = &As[r * 40 + c4 * 4];
      d[0] = (half_t)f.x; d[1] = (half_t)f.y; d[2] = (half_t)f.z; d[3] = (half_t)f.w;
    }
#pragma unroll
    for (int i = 0; i < 4; ++i) {          // W tile: 128 x 32 f32 -> f16
      int idx = tid + 256 * i;
      int r = idx >> 3, c4 = idx & 7;
      float4 f = *(const float4*)&W[(size_t)(bn + r) * Kdim + k0 + c4 * 4];
      half_t* d = &Bs[r * 40 + c4 * 4];
      d[0] = (half_t)f.x; d[1] = (half_t)f.y; d[2] = (half_t)f.z; d[3] = (half_t)f.w;
    }
    __syncthreads();

    v16h af[4], bf[2];
#pragma unroll
    for (int mi = 0; mi < 4; ++mi) af[mi] = load_a_frag(&As[(wm + mi * 16) * 40], 40);
#pragma unroll
    for (int ni = 0; ni < 2; ++ni) bf[ni] = load_b_frag_nk(&Bs[(wn + ni * 16) * 40], 40);
#pragma unroll
    for (int mi = 0; mi < 4; ++mi)
#pragma unroll
      for (int ni = 0; ni < 2; ++ni)
        acc[mi][ni] = wmma16(af[mi], bf[ni], acc[mi][ni]);
    __syncthreads();
  }

#pragma unroll
  for (int mi = 0; mi < 4; ++mi)
#pragma unroll
    for (int ni = 0; ni < 2; ++ni)
#pragma unroll
      for (int r = 0; r < 8; ++r) {
        int row = bm + wm + mi * 16 + hi * 8 + r;
        int col = bn + wn + ni * 16 + l;
        if constexpr (OUT_HALF)
          ((half_t*)Cv)[(size_t)row * Ndim + col] = (half_t)acc[mi][ni][r];
        else
          ((float*)Cv)[(size_t)row * Ndim + col] = acc[mi][ni][r];
      }
}

// ------------------------------ Flash attention -----------------------------
// Grid: (S/64, NH, B); 128 threads = 4 waves; wave w owns query rows 16w..16w+15.
// Q/K/V tiles are f16 in global -> copied with GLOBAL_LOAD_ASYNC_TO_LDS_B128.

__global__ __launch_bounds__(128) void attn_kernel(
    const half_t* __restrict__ Q, const half_t* __restrict__ K,
    const half_t* __restrict__ V, float* __restrict__ O) {
  __shared__ half_t Qs[64 * 72];
  __shared__ half_t Ks[64 * 72];
  __shared__ half_t Vs[64 * 72];
  __shared__ half_t Ps[4][16 * 72];

  const int tid  = threadIdx.x;
  const int lane = tid & 31;
  const int w    = tid >> 5;
  const int l    = lane & 15;
  const int hi   = lane >> 4;

  const int qb = blockIdx.x * 64;
  const int h  = blockIdx.y;
  const int b  = blockIdx.z;

  const size_t rowBase = (size_t)(b * S_) * D_ + h * HD_;

#pragma unroll
  for (int i = 0; i < 4; ++i) {            // Q tile 64x64
    int idx = tid + 128 * i;
    int r = idx >> 3, c8 = idx & 7;
    async_cp16(&Qs[r * 72 + c8 * 8], &Q[rowBase + (size_t)(qb + r) * D_ + c8 * 8]);
  }
  async_wait();
  __syncthreads();

  v16h a_q0 = load_a_frag(&Qs[(16 * w) * 72 + 0],  72);
  v16h a_q1 = load_a_frag(&Qs[(16 * w) * 72 + 32], 72);

  v8f oacc[4] = {};
  float mrow[8], srow[8];
#pragma unroll
  for (int r = 0; r < 8; ++r) { mrow[r] = -1e30f; srow[r] = 0.f; }

  for (int kt = 0; kt < S_ / 64; ++kt) {
    __syncthreads();
#pragma unroll
    for (int i = 0; i < 4; ++i) {          // K/V tiles 64x64
      int idx = tid + 128 * i;
      int r = idx >> 3, c8 = idx & 7;
      size_t g = rowBase + (size_t)(kt * 64 + r) * D_ + c8 * 8;
      async_cp16(&Ks[r * 72 + c8 * 8], &K[g]);
      async_cp16(&Vs[r * 72 + c8 * 8], &V[g]);
    }
    async_wait();
    __syncthreads();

    v8f sacc[4] = {};
#pragma unroll
    for (int ni = 0; ni < 4; ++ni) {       // S = Q K^T
      v16h b0 = load_b_frag_nk(&Ks[(ni * 16) * 72 + 0],  72);
      sacc[ni] = wmma16(a_q0, b0, sacc[ni]);
      v16h b1 = load_b_frag_nk(&Ks[(ni * 16) * 72 + 32], 72);
      sacc[ni] = wmma16(a_q1, b1, sacc[ni]);
    }

    float alpha[8];
#pragma unroll
    for (int r = 0; r < 8; ++r) {          // online softmax per row
      float mx = -1e30f;
#pragma unroll
      for (int ni = 0; ni < 4; ++ni) { sacc[ni][r] *= 0.125f; mx = fmaxf(mx, sacc[ni][r]); }
#pragma unroll
      for (int off = 1; off < 16; off <<= 1) mx = fmaxf(mx, __shfl_xor(mx, off, 32));
      float mn = fmaxf(mrow[r], mx);
      alpha[r] = __expf(mrow[r] - mn);
      mrow[r] = mn;
      float rs = 0.f;
#pragma unroll
      for (int ni = 0; ni < 4; ++ni) { float p = __expf(sacc[ni][r] - mn); sacc[ni][r] = p; rs += p; }
#pragma unroll
      for (int off = 1; off < 16; off <<= 1) rs += __shfl_xor(rs, off, 32);
      srow[r] = srow[r] * alpha[r] + rs;
    }
#pragma unroll
    for (int nt = 0; nt < 4; ++nt)
#pragma unroll
      for (int r = 0; r < 8; ++r) oacc[nt][r] *= alpha[r];

#pragma unroll
    for (int ni = 0; ni < 4; ++ni)         // stage P (f16) for A-operand reload
#pragma unroll
      for (int r = 0; r < 8; ++r)
        Ps[w][(hi * 8 + r) * 72 + ni * 16 + l] = (half_t)sacc[ni][r];
    __syncthreads();

#pragma unroll
    for (int kk = 0; kk < 2; ++kk) {       // O += P V
      v16h ap = load_a_frag(&Ps[w][kk * 32], 72);
#pragma unroll
      for (int nt = 0; nt < 4; ++nt) {
        v16h bv = load_b_frag_kn(&Vs[(kk * 32) * 72 + nt * 16], 72);
        oacc[nt] = wmma16(ap, bv, oacc[nt]);
      }
    }
  }

#pragma unroll
  for (int r = 0; r < 8; ++r) {
    float inv = 1.0f / srow[r];
    int row = qb + 16 * w + hi * 8 + r;
#pragma unroll
    for (int nt = 0; nt < 4; ++nt)
      O[(size_t)(b * S_ + row) * D_ + h * HD_ + nt * 16 + l] = oacc[nt][r] * inv;
  }
}

// ------------------------------ Elementwise ---------------------------------

__global__ __launch_bounds__(256) void rope_kernel(
    half_t* __restrict__ q, half_t* __restrict__ k,
    const float* __restrict__ cosp, const float* __restrict__ sinp) {
  int idx = blockIdx.x * 256 + threadIdx.x;   // one rotation pair per thread
  int d = idx & 31;
  int t = idx >> 5;
  int hh = t & 15; t >>= 4;
  int s = t & 1023;
  int b = t >> 10;
  size_t base = (size_t)(b * S_ + s) * D_ + hh * HD_;
  float c0 = cosp[s * HD_ + d],      s0 = sinp[s * HD_ + d];
  float c1 = cosp[s * HD_ + d + 32], s1 = sinp[s * HD_ + d + 32];
  float q0 = (float)q[base + d], q1 = (float)q[base + d + 32];
  q[base + d]      = (half_t)(q0 * c0 - q1 * s0);
  q[base + d + 32] = (half_t)(q1 * c1 + q0 * s1);
  float k0v = (float)k[base + d], k1 = (float)k[base + d + 32];
  k[base + d]      = (half_t)(k0v * c0 - k1 * s0);
  k[base + d + 32] = (half_t)(k1 * c1 + k0v * s1);
}

// h = rmsnorm(h + d), in place. One row (1024 f32) per block of 256.
__global__ __launch_bounds__(256) void rmsnorm_add_kernel(
    float* __restrict__ h, const float* __restrict__ d) {
  const int row = blockIdx.x;
  const int tid = threadIdx.x;
  const size_t base = (size_t)row * D_ + tid * 4;
  float4 a = *(const float4*)&h[base];
  float4 b = *(const float4*)&d[base];
  a.x += b.x; a.y += b.y; a.z += b.z; a.w += b.w;
  float ss = a.x * a.x + a.y * a.y + a.z * a.z + a.w * a.w;
#pragma unroll
  for (int off = 1; off < 32; off <<= 1) ss += __shfl_xor(ss, off, 32);
  __shared__ float sred[8];
  __shared__ float stot;
  if ((tid & 31) == 0) sred[tid >> 5] = ss;
  __syncthreads();
  if (tid == 0) {
    float t = 0.f;
#pragma unroll
    for (int i = 0; i < 8; ++i) t += sred[i];
    stot = rsqrtf(t * (1.0f / D_) + 1e-5f);
  }
  __syncthreads();
  float sc = stot;
  a.x *= sc; a.y *= sc; a.z *= sc; a.w *= sc;
  *(float4*)&h[base] = a;
}

__global__ __launch_bounds__(256) void silumul_kernel(
    const float* __restrict__ gu, float* __restrict__ act) {
  size_t idx = (size_t)blockIdx.x * 256 + threadIdx.x;
  size_t total = (size_t)M_ * INTER_;
  if (idx >= total) return;
  size_t row = idx / INTER_;
  size_t col = idx - row * INTER_;
  float g = gu[row * (size_t)(2 * INTER_) + col];
  float u = gu[row * (size_t)(2 * INTER_) + INTER_ + col];
  act[idx] = (g / (1.0f + __expf(-g))) * u;
}

__global__ __launch_bounds__(256) void add_inj_kernel(
    float* __restrict__ z, const float* __restrict__ a,
    const float* __restrict__ b, int use_b) {
  size_t i = ((size_t)blockIdx.x * 256 + threadIdx.x) * 4;
  float4 zv = *(const float4*)&z[i];
  float4 av = *(const float4*)&a[i];
  zv.x += av.x; zv.y += av.y; zv.z += av.z; zv.w += av.w;
  if (use_b) {
    float4 bv = *(const float4*)&b[i];
    zv.x += bv.x; zv.y += bv.y; zv.z += bv.z; zv.w += bv.w;
  }
  *(float4*)&z[i] = zv;
}

// ------------------------------ Host orchestration --------------------------

static void run_block(float* h,
                      const float* wq, const float* wk, const float* wv, const float* wo,
                      const float* guw, const float* dnw,
                      half_t* qh, half_t* kh, half_t* vh,
                      float* attn, float* tmp, float* gub, float* actb,
                      const float* cosp, const float* sinp, hipStream_t st) {
  dim3 gD(D_ / 128, M_ / 128);
  gemm_xwt<true ><<<gD, 256, 0, st>>>(h, wq, qh, M_, D_, D_);
  gemm_xwt<true ><<<gD, 256, 0, st>>>(h, wk, kh, M_, D_, D_);
  gemm_xwt<true ><<<gD, 256, 0, st>>>(h, wv, vh, M_, D_, D_);
  rope_kernel<<<(B_ * S_ * NH_ * 32) / 256, 256, 0, st>>>(qh, kh, cosp, sinp);
  attn_kernel<<<dim3(S_ / 64, NH_, B_), 128, 0, st>>>(qh, kh, vh, attn);
  gemm_xwt<false><<<gD, 256, 0, st>>>(attn, wo, tmp, M_, D_, D_);
  rmsnorm_add_kernel<<<M_, 256, 0, st>>>(h, tmp);
  gemm_xwt<false><<<dim3((2 * INTER_) / 128, M_ / 128), 256, 0, st>>>(h, guw, gub, M_, 2 * INTER_, D_);
  size_t tot = (size_t)M_ * INTER_;
  silumul_kernel<<<(unsigned)((tot + 255) / 256), 256, 0, st>>>(gub, actb);
  gemm_xwt<false><<<gD, 256, 0, st>>>(actb, dnw, tmp, M_, D_, INTER_);
  rmsnorm_add_kernel<<<M_, 256, 0, st>>>(h, tmp);
}

static void run_level(float* z, const float* inj_a, const float* inj_b,
                      const float* wq, const float* wk, const float* wv, const float* wo,
                      const float* guw, const float* dnw,
                      half_t* qh, half_t* kh, half_t* vh,
                      float* attn, float* tmp, float* gub, float* actb,
                      const float* cosp, const float* sinp, hipStream_t st) {
  add_inj_kernel<<<(M_ * D_ / 4) / 256, 256, 0, st>>>(z, inj_a, inj_b, inj_b ? 1 : 0);
  const size_t sAtt = (size_t)D_ * D_;
  const size_t sGu  = (size_t)2 * INTER_ * D_;
  const size_t sDn  = (size_t)D_ * INTER_;
  for (int layer = 0; layer < 2; ++layer) {
    run_block(z, wq + layer * sAtt, wk + layer * sAtt, wv + layer * sAtt,
              wo + layer * sAtt, guw + layer * sGu, dnw + layer * sDn,
              qh, kh, vh, attn, tmp, gub, actb, cosp, sinp, st);
  }
}

extern "C" void kernel_launch(void* const* d_in, const int* in_sizes, int n_in,
                              void* d_out, int out_size, void* d_ws, size_t ws_size,
                              hipStream_t stream) {
  (void)in_sizes; (void)n_in; (void)out_size; (void)ws_size;

  const float* zH_in  = (const float*)d_in[0];
  const float* zL_in  = (const float*)d_in[1];
  const float* emb    = (const float*)d_in[2];
  const float* H_wq   = (const float*)d_in[3];
  const float* H_wk   = (const float*)d_in[4];
  const float* H_wv   = (const float*)d_in[5];
  const float* H_wo   = (const float*)d_in[6];
  const float* H_gu   = (const float*)d_in[7];
  const float* H_dn   = (const float*)d_in[8];
  const float* L_wq   = (const float*)d_in[9];
  const float* L_wk   = (const float*)d_in[10];
  const float* L_wv   = (const float*)d_in[11];
  const float* L_wo   = (const float*)d_in[12];
  const float* L_gu   = (const float*)d_in[13];
  const float* L_dn   = (const float*)d_in[14];
  const float* cosp   = (const float*)d_in[15];
  const float* sinp   = (const float*)d_in[16];

  // Workspace layout (bytes)
  const size_t SZ_BSD   = (size_t)M_ * D_ * sizeof(float);       // 16 MB
  const size_t SZ_BSD_H = (size_t)M_ * D_ * sizeof(half_t);      // 8 MB
  const size_t SZ_GU    = (size_t)M_ * 2 * INTER_ * sizeof(float);
  const size_t SZ_ACT   = (size_t)M_ * INTER_ * sizeof(float);

  char* ws = (char*)d_ws;
  size_t off = 0;
  float*  zH   = (float*)(ws + off);  off += SZ_BSD;
  float*  zL   = (float*)(ws + off);  off += SZ_BSD;
  float*  attn = (float*)(ws + off);  off += SZ_BSD;
  float*  tmp  = (float*)(ws + off);  off += SZ_BSD;
  float*  gub  = (float*)(ws + off);  off += SZ_GU;
  float*  actb = (float*)(ws + off);  off += SZ_ACT;
  half_t* qh   = (half_t*)(ws + off); off += SZ_BSD_H;
  half_t* kh   = (half_t*)(ws + off); off += SZ_BSD_H;
  half_t* vh   = (half_t*)(ws + off); off += SZ_BSD_H;

  (void)hipMemcpyAsync(zH, zH_in, SZ_BSD, hipMemcpyDeviceToDevice, stream);
  (void)hipMemcpyAsync(zL, zL_in, SZ_BSD, hipMemcpyDeviceToDevice, stream);

  for (int hc = 0; hc < 2; ++hc) {
    for (int lc = 0; lc < 2; ++lc) {
      run_level(zL, zH, emb, L_wq, L_wk, L_wv, L_wo, L_gu, L_dn,
                qh, kh, vh, attn, tmp, gub, actb, cosp, sinp, stream);
    }
    run_level(zH, zL, nullptr, H_wq, H_wk, H_wv, H_wo, H_gu, H_dn,
              qh, kh, vh, attn, tmp, gub, actb, cosp, sinp, stream);
  }

  (void)hipMemcpyAsync(d_out, zH, SZ_BSD, hipMemcpyDeviceToDevice, stream);
}
